// EdgeAwareEncoder_82626580840483
// MI455X (gfx1250) — compile-verified, matrix operand order checked by reading, not run on hardware
//
#include <hip/hip_runtime.h>

// ---------------------------------------------------------------------------
// GINE VGAE encoder for MI455X (gfx1250, wave32, WMMA).
//
//   edge:  e = edge_attr @ We + be ; msg = relu(x[src]+e) ; aggr[dst] += msg
//   node:  out = Lin2(relu(Lin1(x + aggr)))  (+ outer relu / clip)
//
// Matmuls: v_wmma_f32_16x16x32_bf16 (bf16 A/B, f32 accum).
// Gathers/atomics L2-resident (x/h/aggr fit in 192MB L2); hot-path
// addressing is saddr + 32-bit voffset with scale_offset. Edge indices are
// loaded as int4 vectors (L0-broadcast), gathers batched ahead of atomics.
// conv_mu/conv_logstd edge stages fused (shared A tiles + h[src] gathers).
// ---------------------------------------------------------------------------

typedef __attribute__((ext_vector_type(16))) __bf16 bfvec16;
typedef __attribute__((ext_vector_type(8)))  float  fvec8;

__device__ __forceinline__ __bf16 tobf(float f) {
  union { float f; unsigned u; } cv; cv.f = f;
  unsigned u = cv.u + 0x7FFFu + ((cv.u >> 16) & 1u);   // RNE to bf16
  union { unsigned short s; __bf16 b; } o;
  o.s = (unsigned short)(u >> 16);
  return o.b;
}
__device__ __forceinline__ __bf16 bf0() {
  union { unsigned short s; __bf16 b; } o; o.s = 0; return o.b;
}
// Pack two floats into two RNE bf16s in one dword (lo -> [15:0], hi -> [31:16]).
__device__ __forceinline__ unsigned pack_bf16(float lo, float hi) {
  union { float f; unsigned u; } a, b; a.f = lo; b.f = hi;
  const unsigned ua = a.u + 0x7FFFu + ((a.u >> 16) & 1u);
  const unsigned ub = b.u + 0x7FFFu + ((b.u >> 16) & 1u);
  return (ua >> 16) | (ub & 0xFFFF0000u);
}

// ---------------------------------------------------------------------------
__global__ void zero_kernel(float4* __restrict__ p, size_t n4) {
  size_t i = (size_t)blockIdx.x * blockDim.x + threadIdx.x;
  const size_t stride = (size_t)gridDim.x * blockDim.x;
  const float4 z = make_float4(0.f, 0.f, 0.f, 0.f);
  for (; i < n4; i += stride) p[i] = z;
}

// Load 8 consecutive edge indices (this lane's C rows) as two int4 and
// convert to row offsets (idx * DOUT). base16 must be 16B-aligned.
template<int DOUT>
__device__ __forceinline__ void load_rows8(const int* __restrict__ idx,
                                           int ebase, int khalf,
                                           unsigned row[8]) {
  const int4* p = (const int4*)(idx + ebase) + khalf * 2;
  const int4 a = p[0];
  const int4 b = p[1];
  row[0] = (unsigned)a.x * DOUT; row[1] = (unsigned)a.y * DOUT;
  row[2] = (unsigned)a.z * DOUT; row[3] = (unsigned)a.w * DOUT;
  row[4] = (unsigned)b.x * DOUT; row[5] = (unsigned)b.y * DOUT;
  row[6] = (unsigned)b.z * DOUT; row[7] = (unsigned)b.w * DOUT;
}

// ---------------------------------------------------------------------------
// Edge stage (conv1): one wave = 16 edges.
//   A (16x32 bf16): rows = edges, K = 16 attrs (K 16..31 zero padded);
//     lanes 0-15 hold K0..7 (elems 0-7), lanes 16-31 hold K8..15.
//   B: We tile preloaded in registers, same K packing, N = lane&15.
//   C: VGPR r -> edge row (r + 8*(lane>=16)), lane&15 -> channel.
// ---------------------------------------------------------------------------
template<int DOUT>
__global__ __launch_bounds__(256)
void gine_edge_kernel(const float* __restrict__ xin,   // [N, DOUT]
                      const int*   __restrict__ src,
                      const int*   __restrict__ dst,
                      const float* __restrict__ eattr, // [E, 16]
                      const float* __restrict__ We,    // [16, DOUT]
                      const float* __restrict__ be,    // [DOUT]
                      float*       __restrict__ aggr,  // [N, DOUT]
                      int nEdges) {
  constexpr int NT = DOUT / 16;
  const int lane  = threadIdx.x & 31;
  const int wave  = threadIdx.x >> 5;
  const int khalf = lane >> 4;
  const int nl    = lane & 15;

  bfvec16 btile[NT];
  float   bias[NT];
#pragma unroll
  for (int t = 0; t < NT; ++t) {
    const int ch = t * 16 + nl;
    bfvec16 b;
#pragma unroll
    for (int e = 0; e < 8; ++e) {
      b[e]     = tobf(We[(khalf * 8 + e) * DOUT + ch]);
      b[8 + e] = bf0();
    }
    btile[t] = b;
    bias[t]  = be[ch];
  }

  const int tilesTotal = nEdges >> 4;
  const int wavesTotal = gridDim.x * (blockDim.x >> 5);
  const int waveId     = blockIdx.x * (blockDim.x >> 5) + wave;

  for (int tile = waveId; tile < tilesTotal; tile += wavesTotal) {
    const int ebase = tile << 4;

    // Edge indices: 4x global_load_b128, L0-broadcast across lanes.
    unsigned srow[8], drow[8];
    load_rows8<DOUT>(src, ebase, khalf, srow);
    load_rows8<DOUT>(dst, ebase, khalf, drow);

    const float* ap = eattr + (size_t)(ebase + nl) * 16 + khalf * 8;
    bfvec16 a;
#pragma unroll
    for (int e = 0; e < 8; ++e) { a[e] = tobf(ap[e]); a[8 + e] = bf0(); }

#pragma unroll
    for (int t = 0; t < NT; ++t) {
      fvec8 c = {};
      c = __builtin_amdgcn_wmma_f32_16x16x32_bf16(false, a, false, btile[t],
                                                  (short)0, c, false, false);
      const unsigned ch = t * 16 + nl;
      float g[8];
#pragma unroll
      for (int r = 0; r < 8; ++r) g[r] = xin[srow[r] + ch];   // batched gathers
#pragma unroll
      for (int r = 0; r < 8; ++r) {
        float v = c[r] + bias[t] + g[r];
        v = v > 0.0f ? v : 0.0f;
        atomicAdd(&aggr[drow[r] + ch], v);
      }
    }
  }
}

// ---------------------------------------------------------------------------
// Fused edge stage for conv_mu + conv_logstd (same A tiles, shared h[src]
// gathers, two weight sets, two aggregation targets).
// ---------------------------------------------------------------------------
template<int DOUT>
__global__ __launch_bounds__(256)
void gine_edge_dual_kernel(const float* __restrict__ xin,
                           const int*   __restrict__ src,
                           const int*   __restrict__ dst,
                           const float* __restrict__ eattr,
                           const float* __restrict__ We2,
                           const float* __restrict__ be2,
                           const float* __restrict__ We3,
                           const float* __restrict__ be3,
                           float*       __restrict__ aggr2,
                           float*       __restrict__ aggr3,
                           int nEdges) {
  constexpr int NT = DOUT / 16;
  const int lane  = threadIdx.x & 31;
  const int wave  = threadIdx.x >> 5;
  const int khalf = lane >> 4;
  const int nl    = lane & 15;

  bfvec16 b2t[NT], b3t[NT];
  float   bias2[NT], bias3[NT];
#pragma unroll
  for (int t = 0; t < NT; ++t) {
    const int ch = t * 16 + nl;
    bfvec16 b2, b3;
#pragma unroll
    for (int e = 0; e < 8; ++e) {
      const int k = khalf * 8 + e;
      b2[e] = tobf(We2[k * DOUT + ch]);  b2[8 + e] = bf0();
      b3[e] = tobf(We3[k * DOUT + ch]);  b3[8 + e] = bf0();
    }
    b2t[t] = b2;  b3t[t] = b3;
    bias2[t] = be2[ch];  bias3[t] = be3[ch];
  }

  const int tilesTotal = nEdges >> 4;
  const int wavesTotal = gridDim.x * (blockDim.x >> 5);
  const int waveId     = blockIdx.x * (blockDim.x >> 5) + wave;

  for (int tile = waveId; tile < tilesTotal; tile += wavesTotal) {
    const int ebase = tile << 4;

    unsigned srow[8], drow[8];
    load_rows8<DOUT>(src, ebase, khalf, srow);
    load_rows8<DOUT>(dst, ebase, khalf, drow);

    const float* ap = eattr + (size_t)(ebase + nl) * 16 + khalf * 8;
    bfvec16 a;
#pragma unroll
    for (int e = 0; e < 8; ++e) { a[e] = tobf(ap[e]); a[8 + e] = bf0(); }

#pragma unroll
    for (int t = 0; t < NT; ++t) {
      fvec8 c2 = {}, c3 = {};
      c2 = __builtin_amdgcn_wmma_f32_16x16x32_bf16(false, a, false, b2t[t],
                                                   (short)0, c2, false, false);
      c3 = __builtin_amdgcn_wmma_f32_16x16x32_bf16(false, a, false, b3t[t],
                                                   (short)0, c3, false, false);
      const unsigned ch = t * 16 + nl;
      float g[8];
#pragma unroll
      for (int r = 0; r < 8; ++r) g[r] = xin[srow[r] + ch];   // shared gathers
#pragma unroll
      for (int r = 0; r < 8; ++r) {
        float v2 = c2[r] + bias2[t] + g[r];  v2 = v2 > 0.0f ? v2 : 0.0f;
        float v3 = c3[r] + bias3[t] + g[r];  v3 = v3 > 0.0f ? v3 : 0.0f;
        atomicAdd(&aggr2[drow[r] + ch], v2);
        atomicAdd(&aggr3[drow[r] + ch], v3);
      }
    }
  }
}

// Scalar fallbacks for edges beyond the last full 16-edge tile.
template<int DOUT>
__global__ void gine_edge_tail_kernel(const float* __restrict__ xin,
                                      const int* __restrict__ src,
                                      const int* __restrict__ dst,
                                      const float* __restrict__ eattr,
                                      const float* __restrict__ We,
                                      const float* __restrict__ be,
                                      float* __restrict__ aggr,
                                      int estart, int nEdges) {
  const int idx = blockIdx.x * blockDim.x + threadIdx.x;
  const int e = estart + idx / DOUT;
  const int ch = idx % DOUT;
  if (e >= nEdges) return;
  float acc = be[ch];
  for (int k = 0; k < 16; ++k) acc += eattr[(size_t)e * 16 + k] * We[k * DOUT + ch];
  float v = acc + xin[(unsigned)src[e] * DOUT + ch];
  v = v > 0.0f ? v : 0.0f;
  atomicAdd(&aggr[(unsigned)dst[e] * DOUT + ch], v);
}

template<int DOUT>
__global__ void gine_edge_dual_tail_kernel(const float* __restrict__ xin,
                                           const int* __restrict__ src,
                                           const int* __restrict__ dst,
                                           const float* __restrict__ eattr,
                                           const float* __restrict__ We2,
                                           const float* __restrict__ be2,
                                           const float* __restrict__ We3,
                                           const float* __restrict__ be3,
                                           float* __restrict__ aggr2,
                                           float* __restrict__ aggr3,
                                           int estart, int nEdges) {
  const int idx = blockIdx.x * blockDim.x + threadIdx.x;
  const int e = estart + idx / DOUT;
  const int ch = idx % DOUT;
  if (e >= nEdges) return;
  float a2 = be2[ch], a3 = be3[ch];
  for (int k = 0; k < 16; ++k) {
    const float ea = eattr[(size_t)e * 16 + k];
    a2 += ea * We2[k * DOUT + ch];
    a3 += ea * We3[k * DOUT + ch];
  }
  const float g = xin[(unsigned)src[e] * DOUT + ch];
  float v2 = a2 + g;  v2 = v2 > 0.0f ? v2 : 0.0f;
  float v3 = a3 + g;  v3 = v3 > 0.0f ? v3 : 0.0f;
  atomicAdd(&aggr2[(unsigned)dst[e] * DOUT + ch], v2);
  atomicAdd(&aggr3[(unsigned)dst[e] * DOUT + ch], v3);
}

// ---------------------------------------------------------------------------
// Node MLP stage: block = 8 waves = 128 nodes. Two WMMA layers through LDS.
// z = x + aggr staged to LDS as packed bf16 pairs (rows padded +8), weights
// staged transposed so A and B tiles read as aligned dword pairs matching
// the 16-bit A layout. MODE: 0 = relu (h), 1 = identity (mu), 2 = clip.
// ---------------------------------------------------------------------------
template<int DIN, int MODE>
__global__ __launch_bounds__(256)
void gine_node_mlp_kernel(const float* __restrict__ zin,   // [N, DIN]
                          const float* __restrict__ aggr,  // [N, DIN]
                          const float* __restrict__ Wa,    // [DIN, 64]
                          const float* __restrict__ ba,    // [64]
                          const float* __restrict__ Wb,    // [64, 64]
                          const float* __restrict__ bb,    // [64]
                          float*       __restrict__ out,   // [N, 64]
                          int nNodes) {
  constexpr int DH = 64;
  constexpr int PAD = 8;
  __shared__ __bf16 zb [128][DIN + PAD];
  __shared__ __bf16 WaT[DH][DIN + PAD];
  __shared__ __bf16 WbT[DH][DH + PAD];
  __shared__ __bf16 t1 [128][DH + PAD];
  __shared__ float  baS[DH], bbS[DH];

  const int tid   = threadIdx.x;
  const int lane  = tid & 31;
  const int wave  = tid >> 5;
  const int khalf = lane >> 4;
  const int nl    = lane & 15;
  const int nodeRow = wave * 16;

  constexpr int DW  = DIN / 2;
  constexpr int DW2 = DH / 2;
  for (int i = tid; i < DH * DW; i += 256) {
    const int ch = i / DW, k = (i % DW) * 2;
    *(unsigned*)&WaT[ch][k] =
        pack_bf16(Wa[(unsigned)k * DH + ch], Wa[(unsigned)(k + 1) * DH + ch]);
  }
  for (int i = tid; i < DH * DW2; i += 256) {
    const int ch = i / DW2, k = (i % DW2) * 2;
    *(unsigned*)&WbT[ch][k] =
        pack_bf16(Wb[(unsigned)k * DH + ch], Wb[(unsigned)(k + 1) * DH + ch]);
  }
  if (tid < DH) { baS[tid] = ba[tid]; bbS[tid] = bb[tid]; }

  union BV { bfvec16 v; unsigned u[8]; };

  const int nBlk = (nNodes + 127) / 128;
  for (int blk = blockIdx.x; blk < nBlk; blk += gridDim.x) {
    const int base = blk * 128;

    // Stage z = zin + aggr as packed bf16 pairs.
    for (int i = tid; i < 128 * DW; i += 256) {
      const int r = i / DW, k = (i % DW) * 2;
      int gn = base + r; if (gn >= nNodes) gn = nNodes - 1;  // clamp; no store later
      const unsigned zo = (unsigned)gn * DIN + k;
      *(unsigned*)&zb[r][k] =
          pack_bf16(zin[zo] + aggr[zo], zin[zo + 1] + aggr[zo + 1]);
    }
    __syncthreads();

    // ----- Layer 1: t1 = relu(z @ Wa + ba) -----
    const unsigned* zr = (const unsigned*)&zb[nodeRow + nl][0];
#pragma unroll
    for (int t = 0; t < DH / 16; ++t) {
      const unsigned* wr = (const unsigned*)&WaT[t * 16 + nl][0];
      fvec8 c = {};
#pragma unroll
      for (int kb = 0; kb < DIN / 32; ++kb) {
        BV av, bv;
#pragma unroll
        for (int i = 0; i < 8; ++i) {
          // dword index implementing A-layout K packing.
          const int dw = (i & 3) + ((i >> 2) << 3) + (khalf << 2) + kb * 16;
          av.u[i] = zr[dw];
          bv.u[i] = wr[dw];
        }
        c = __builtin_amdgcn_wmma_f32_16x16x32_bf16(false, av.v, false, bv.v,
                                                    (short)0, c, false, false);
      }
#pragma unroll
      for (int r = 0; r < 8; ++r) {
        float v = c[r] + baS[t * 16 + nl];
        v = v > 0.0f ? v : 0.0f;
        t1[nodeRow + r + (khalf << 3)][t * 16 + nl] = tobf(v);
      }
    }
    __syncthreads();

    // ----- Layer 2: out = t1 @ Wb + bb (+ epilogue) -----
    const unsigned* tr = (const unsigned*)&t1[nodeRow + nl][0];
#pragma unroll
    for (int t = 0; t < DH / 16; ++t) {
      const unsigned* wr = (const unsigned*)&WbT[t * 16 + nl][0];
      fvec8 c = {};
#pragma unroll
      for (int kb = 0; kb < DH / 32; ++kb) {
        BV av, bv;
#pragma unroll
        for (int i = 0; i < 8; ++i) {
          const int dw = (i & 3) + ((i >> 2) << 3) + (khalf << 2) + kb * 16;
          av.u[i] = tr[dw];
          bv.u[i] = wr[dw];
        }
        c = __builtin_amdgcn_wmma_f32_16x16x32_bf16(false, av.v, false, bv.v,
                                                    (short)0, c, false, false);
      }
#pragma unroll
      for (int r = 0; r < 8; ++r) {
        const int row = nodeRow + r + (khalf << 3);
        const int gn = base + row;
        float v = c[r] + bbS[t * 16 + nl];
        if (MODE == 0) v = v > 0.0f ? v : 0.0f;
        if (MODE == 2) v = fminf(fmaxf(v, -10.0f), 10.0f);
        if (gn < nNodes) out[(unsigned)gn * DH + t * 16 + nl] = v;
      }
    }
    __syncthreads();
  }
}

// ---------------------------------------------------------------------------
extern "C" void kernel_launch(void* const* d_in, const int* in_sizes, int n_in,
                              void* d_out, int out_size, void* d_ws, size_t ws_size,
                              hipStream_t stream) {
  const float* x   = (const float*)d_in[0];
  const int*   ei  = (const int*)d_in[1];
  const float* ea  = (const float*)d_in[2];
  const float *We1 = (const float*)d_in[3],  *be1 = (const float*)d_in[4];
  const float *W1a = (const float*)d_in[5],  *b1a = (const float*)d_in[6];
  const float *W1b = (const float*)d_in[7],  *b1b = (const float*)d_in[8];
  const float *We2 = (const float*)d_in[9],  *be2 = (const float*)d_in[10];
  const float *W2a = (const float*)d_in[11], *b2a = (const float*)d_in[12];
  const float *W2b = (const float*)d_in[13], *b2b = (const float*)d_in[14];
  const float *We3 = (const float*)d_in[15], *be3 = (const float*)d_in[16];
  const float *W3a = (const float*)d_in[17], *b3a = (const float*)d_in[18];
  const float *W3b = (const float*)d_in[19], *b3b = (const float*)d_in[20];

  const int N = in_sizes[0] / 128;
  const int E = in_sizes[1] / 2;
  const int* src = ei;
  const int* dst = ei + E;

  // Workspace: aggr1 [N,128] (reused as aggr2+aggr3, each [N,64]); h [N,64].
  float* aggr1 = (float*)d_ws;
  float* h     = aggr1 + (size_t)N * 128;
  float* aggr2 = aggr1;
  float* aggr3 = aggr1 + (size_t)N * 64;
  float* mu    = (float*)d_out;
  float* lstd  = mu + (size_t)N * 64;

  // Vector index loads need 16B alignment of src/dst tiles: requires E % 4 == 0
  // (holds here: E = 1.6M). Otherwise fall back to full-range scalar tail.
  const bool vecOK = (E % 4) == 0;
  const int tiles     = vecOK ? (E >> 4) : 0;
  const int tailStart = tiles << 4;
  const int tailCount = E - tailStart;

  int edgeGrid = (tiles + 7) / 8;           // one 16-edge tile per wave
  if (edgeGrid < 1) edgeGrid = 1;
  if (edgeGrid > 16384) edgeGrid = 16384;   // grid-stride covers the rest
  const int mlpGrid = (N + 127) / 128;
  const dim3 blk(256);

  // ---- conv1: x[N,128] -> h[N,64] ----
  zero_kernel<<<2048, 256, 0, stream>>>((float4*)aggr1, (size_t)N * 32);
  if (tiles)
    gine_edge_kernel<128><<<edgeGrid, blk, 0, stream>>>(x, src, dst, ea, We1, be1, aggr1, E);
  if (tailCount)
    gine_edge_tail_kernel<128><<<((long)tailCount * 128 + 255) / 256, 256, 0, stream>>>(
        x, src, dst, ea, We1, be1, aggr1, tailStart, E);
  gine_node_mlp_kernel<128, 0><<<mlpGrid, blk, 0, stream>>>(x, aggr1, W1a, b1a, W1b, b1b, h, N);

  // ---- conv_mu + conv_logstd (fused edge stage): h[N,64] -> aggr2, aggr3 ----
  zero_kernel<<<2048, 256, 0, stream>>>((float4*)aggr1, (size_t)N * 32);  // zeros both
  if (tiles)
    gine_edge_dual_kernel<64><<<edgeGrid, blk, 0, stream>>>(
        h, src, dst, ea, We2, be2, We3, be3, aggr2, aggr3, E);
  if (tailCount)
    gine_edge_dual_tail_kernel<64><<<((long)tailCount * 64 + 255) / 256, 256, 0, stream>>>(
        h, src, dst, ea, We2, be2, We3, be3, aggr2, aggr3, tailStart, E);

  gine_node_mlp_kernel<64, 1><<<mlpGrid, blk, 0, stream>>>(h, aggr2, W2a, b2a, W2b, b2b, mu, N);
  gine_node_mlp_kernel<64, 2><<<mlpGrid, blk, 0, stream>>>(h, aggr3, W3a, b3a, W3b, b3b, lstd, N);
}
